// GAE_NP_58248346469023
// MI455X (gfx1250) — matrix operand order repaired and checked by reading, not compiled, not run on hardware
//
#include <hip/hip_runtime.h>
#include <hip/hip_bf16.h>

// Problem constants (from reference)
#define N_NODES 8192
#define D_IN    512
#define H_DIM   256
#define L_DIM   64
#define N_EDGES 262144

typedef __attribute__((ext_vector_type(16))) __bf16         v16bf;
typedef __attribute__((ext_vector_type(16))) unsigned short v16s;
typedef __attribute__((ext_vector_type(8)))  unsigned int   v8u;
typedef __attribute__((ext_vector_type(8)))  float          v8f;

__device__ __forceinline__ unsigned short f32_to_bf16_bits(float f) {
  unsigned u = __builtin_bit_cast(unsigned, f);
  u += 0x7FFFu + ((u >> 16) & 1u);          // round-to-nearest-even
  return (unsigned short)(u >> 16);
}

// ---------------------------------------------------------------------------
// Dense GEMM: C[M,N] = (RELU_A ? relu(A) : A) @ B, A:[M,K] f32, B:[K,N] f32.
// One wave computes one 16x16 tile via v_wmma_f32_16x16x32_bf16, K-loop of 32.
// ---------------------------------------------------------------------------
template <int KDIM, int NDIM, bool RELU_A>
__global__ void gemm_bf16_wmma(const float* __restrict__ A,
                               const float* __restrict__ B,
                               float* __restrict__ C,
                               int tilesN) {
  const int wave   = threadIdx.x >> 5;
  const int lane   = threadIdx.x & 31;
  const int tile   = blockIdx.x * (blockDim.x >> 5) + wave;
  const int tileM  = tile / tilesN;
  const int tileN  = tile % tilesN;
  const int laneLo = lane & 15;
  const int laneHi = lane >> 4;
  const int rowA   = tileM * 16 + laneLo;   // A fragment row (M)
  const int colB   = tileN * 16 + laneLo;   // B fragment col (N)

  v8f acc = {};
  for (int k0 = 0; k0 < KDIM; k0 += 32) {
    v16s au, bu;
#pragma unroll
    for (int v = 0; v < 8; ++v) {
      // A layout (16-bit A 16x32): VGPR v holds K pair; halves offset by 8.
      int ka = k0 + ((v >> 2) << 4) + (laneHi << 3) + ((v & 3) << 1);
      float a0 = A[rowA * KDIM + ka];
      float a1 = A[rowA * KDIM + ka + 1];
      if (RELU_A) { a0 = fmaxf(a0, 0.0f); a1 = fmaxf(a1, 0.0f); }
      au[2 * v]     = f32_to_bf16_bits(a0);
      au[2 * v + 1] = f32_to_bf16_bits(a1);
      // B layout (16-bit B 32x16): VGPR v holds K pair; halves offset by 8.
      int kb = k0 + ((v >> 2) << 4) + (laneHi << 3) + ((v & 3) << 1);
      bu[2 * v]     = f32_to_bf16_bits(B[kb * NDIM + colB]);
      bu[2 * v + 1] = f32_to_bf16_bits(B[(kb + 1) * NDIM + colB]);
    }
    acc = __builtin_amdgcn_wmma_f32_16x16x32_bf16(
        false, __builtin_bit_cast(v16bf, au),
        false, __builtin_bit_cast(v16bf, bu),
        (short)0, acc, false, false);
  }
#pragma unroll
  for (int r = 0; r < 8; ++r) {             // C/D layout: m = r + 8*laneHi
    int m = tileM * 16 + laneHi * 8 + r;
    int n = tileN * 16 + laneLo;
    C[m * NDIM + n] = acc[r];
  }
}

// ---------------------------------------------------------------------------
// SpMM scatter: dst[row[e], f] += w[e] * src[col[e], f]  (hardware f32 atomics)
// Thread (e,f) mapping keeps lanes coalesced across features of one edge.
// ---------------------------------------------------------------------------
template <int F>
__global__ void spmm_scatter(const int* __restrict__ erow,
                             const int* __restrict__ ecol,
                             const float* __restrict__ ew,
                             const float* __restrict__ src,
                             float* __restrict__ dst) {
  unsigned gid = blockIdx.x * blockDim.x + threadIdx.x;
  unsigned e = gid / F;                     // F is a power of two -> shifts
  unsigned f = gid % F;
  int r = erow[e];
  int c = ecol[e];
  float v = ew[e] * src[c * F + f];
  __hip_atomic_fetch_add(&dst[r * F + f], v,
                         __ATOMIC_RELAXED, __HIP_MEMORY_SCOPE_AGENT);
}

// ---------------------------------------------------------------------------
// z_bf16 = bf16(relu(z_pre))
// ---------------------------------------------------------------------------
__global__ void relu_to_bf16(const float* __restrict__ in,
                             unsigned short* __restrict__ out) {
  int i = blockIdx.x * blockDim.x + threadIdx.x;
  out[i] = f32_to_bf16_bits(fmaxf(in[i], 0.0f));
}

// ---------------------------------------------------------------------------
// Decode: out = (sigmoid(z @ z^T) + FUDGE) * (1 - 2*FUDGE), z: [8192,64] bf16.
// Store-bandwidth-bound: 2 WMMAs per 16x16 tile; z (1 MB bf16) is L2-resident.
// Fast epilogue: native v_tanh_f32 (gfx1250) or raw v_rcp_f32, and
// non-temporal output stores so the 268 MB result doesn't evict z from L2.
// zb is viewed as packed bf16 pairs (uint), row stride = 32 uints.
// ---------------------------------------------------------------------------
__global__ void decode_wmma(const unsigned int* __restrict__ zb,
                            float* __restrict__ out) {
  const int wave   = threadIdx.x >> 5;
  const int lane   = threadIdx.x & 31;
  const int tile   = blockIdx.x * (blockDim.x >> 5) + wave;
  const int tileM  = tile >> 9;             // 512 tile-columns
  const int tileN  = tile & 511;
  const int laneLo = lane & 15;
  const int laneHi = lane >> 4;
  const int rowA   = tileM * 16 + laneLo;   // row of z
  const int rowB   = tileN * 16 + laneLo;   // B[k][n] = z[n][k] -> row n of z

  v8f acc = {};
#pragma unroll
  for (int k0 = 0; k0 < L_DIM; k0 += 32) {
    v8u au, bu;
#pragma unroll
    for (int v = 0; v < 8; ++v) {
      int ka = k0 + ((v >> 2) << 4) + (laneHi << 3) + ((v & 3) << 1);
      au[v] = zb[(rowA << 5) + (ka >> 1)];  // pair (ka, ka+1), ka even
      int kb = k0 + ((v >> 2) << 4) + (laneHi << 3) + ((v & 3) << 1);
      bu[v] = zb[(rowB << 5) + (kb >> 1)];
    }
    acc = __builtin_amdgcn_wmma_f32_16x16x32_bf16(
        false, __builtin_bit_cast(v16bf, au),
        false, __builtin_bit_cast(v16bf, bu),
        (short)0, acc, false, false);
  }
  const float scale = 1.0f - 2.0f * 1e-7f;  // (p + FUDGE)*scale = p*scale + bias
  const float bias  = 1e-7f * scale;
#pragma unroll
  for (int r = 0; r < 8; ++r) {
    int m = tileM * 16 + laneHi * 8 + r;
    int n = tileN * 16 + laneLo;
    float s = acc[r];
#if __has_builtin(__builtin_amdgcn_tanhf)
    // sigmoid(s) = 0.5 + 0.5*tanh(s/2): single v_tanh_f32 + FMA (gfx1250)
    float p = fmaf(0.5f, __builtin_amdgcn_tanhf(0.5f * s), 0.5f);
#else
    // v_exp_f32 + raw v_rcp_f32 (skip IEEE div refinement)
    float p = __builtin_amdgcn_rcpf(1.0f + __expf(-s));
#endif
    float val = fmaf(p, scale, bias);
    __builtin_nontemporal_store(val, &out[(unsigned)m * (unsigned)N_NODES +
                                          (unsigned)n]);
  }
}

// ---------------------------------------------------------------------------
// Host-side launch (graph-capture safe: only async stream ops)
// ---------------------------------------------------------------------------
extern "C" void kernel_launch(void* const* d_in, const int* in_sizes, int n_in,
                              void* d_out, int out_size, void* d_ws, size_t ws_size,
                              hipStream_t stream) {
  (void)in_sizes; (void)n_in; (void)out_size; (void)ws_size;

  const float* x   = (const float*)d_in[0];
  const int*   er  = (const int*)d_in[1];
  const int*   ec  = (const int*)d_in[2];
  const float* ew  = (const float*)d_in[3];
  const float* W1  = (const float*)d_in[4];
  const float* W2  = (const float*)d_in[5];
  float*       out = (float*)d_out;

  // Workspace layout (21 MB total)
  char* ws = (char*)d_ws;
  float*          xw1  = (float*)(ws);                        //  8 MB [8192,256]
  float*          hpre = (float*)(ws + (size_t)( 8u << 20));  //  8 MB [8192,256]
  float*          hw2  = (float*)(ws + (size_t)(16u << 20));  //  2 MB [8192,64]
  float*          zpre = (float*)(ws + (size_t)(18u << 20));  //  2 MB [8192,64]
  unsigned short* zbf  = (unsigned short*)(ws + (size_t)(20u << 20)); // 1 MB

  hipMemsetAsync(hpre, 0, (size_t)N_NODES * H_DIM * sizeof(float), stream);
  hipMemsetAsync(zpre, 0, (size_t)N_NODES * L_DIM * sizeof(float), stream);

  // GEMM1: xw1 = x @ W1   (M=8192, K=512, N=256) -> 512*16 = 8192 tiles
  gemm_bf16_wmma<D_IN, H_DIM, false>
      <<<8192 / 4, 128, 0, stream>>>(x, W1, xw1, H_DIM / 16);

  // SpMM1: hpre[row,:] += w * xw1[col,:]   (E*256 = 67M fused scatter-adds)
  spmm_scatter<H_DIM>
      <<<(N_EDGES * (unsigned)H_DIM) / 256u, 256, 0, stream>>>(er, ec, ew, xw1, hpre);

  // GEMM2: hw2 = relu(hpre) @ W2  (M=8192, K=256, N=64) -> 512*4 = 2048 tiles
  gemm_bf16_wmma<H_DIM, L_DIM, true>
      <<<2048 / 4, 128, 0, stream>>>(hpre, W2, hw2, L_DIM / 16);

  // SpMM2: zpre[row,:] += w * hw2[col,:]
  spmm_scatter<L_DIM>
      <<<(N_EDGES * (unsigned)L_DIM) / 256u, 256, 0, stream>>>(er, ec, ew, hw2, zpre);

  // z (bf16) = relu(zpre)
  relu_to_bf16<<<(N_NODES * L_DIM) / 256, 256, 0, stream>>>(zpre, zbf);

  // Decode: 512*512 = 262144 tiles, 8 waves per 256-thread block
  decode_wmma<<<262144 / 8, 256, 0, stream>>>((const unsigned int*)zbf, out);
}
